// PIEv2Loss_34737695490217
// MI455X (gfx1250) — matrix operand order repaired and checked by reading, not compiled, not run on hardware
//
#include <hip/hip_runtime.h>
#include <hip/hip_bf16.h>
#include <math.h>

#define BATCH   1024
#define EMBED   512
#define NCLS    100000
#define MARGIN  0.3f
#define EPS_S   0.1f

#define CT  128                      // class tile per block
#define RT  64                       // row tile per block
#define KC  256                      // K chunk staged in LDS
#define XSS 264                      // Xs row stride (bf16 elems), bank-conflict free
#define WSS 264
#define LTS 132                      // logits LDS stride (floats)
#define NT  ((NCLS + CT - 1) / CT)   // 782 class tiles

typedef __bf16 bf16_t;
typedef bf16_t bf16x16 __attribute__((ext_vector_type(16)));
typedef bf16_t bf16x2  __attribute__((ext_vector_type(2)));
typedef float  f32x8   __attribute__((ext_vector_type(8)));
typedef float  f32x2   __attribute__((ext_vector_type(2)));

// ---------------- init: dist_ap = 0 (as int bits), dist_an = +inf ----------------
__global__ __launch_bounds__(256) void k_init(int* ap, int* an, float* lab) {
  int i = blockIdx.x * blockDim.x + threadIdx.x;
  if (i < BATCH) { ap[i] = 0; an[i] = 0x7f800000; lab[i] = 0.f; }
}

// ---------------- per-row squared norms ----------------
__global__ __launch_bounds__(128) void k_sq(const float* __restrict__ x, float* __restrict__ sq) {
  __shared__ float red[128];
  const int row = blockIdx.x;
  float4 v = ((const float4*)(x + (size_t)row * EMBED))[threadIdx.x];
  red[threadIdx.x] = v.x * v.x + v.y * v.y + v.z * v.z + v.w * v.w;
  __syncthreads();
  for (int s = 64; s > 0; s >>= 1) {
    if (threadIdx.x < s) red[threadIdx.x] += red[threadIdx.x + s];
    __syncthreads();
  }
  if (threadIdx.x == 0) sq[row] = red[0];
}

// ---------------- Gram + batch-hard mining, exact f32 WMMA 16x16x4 ----------------
__global__ __launch_bounds__(32) void k_gram(const float* __restrict__ x,
                                             const float* __restrict__ sq,
                                             const int*   __restrict__ labels,
                                             int* __restrict__ ap, int* __restrict__ an) {
  const int iBase = blockIdx.x * 16;
  const int jBase = blockIdx.y * 16;
  const int lane  = threadIdx.x;
  const int g = lane >> 4, m = lane & 15;   // A: M=m, B: N=m
  f32x8 acc = {};
  const float* xi = x + (size_t)(iBase + m) * EMBED;
  const float* xj = x + (size_t)(jBase + m) * EMBED;
  #pragma unroll 4
  for (int kb = 0; kb < EMBED; kb += 4) {
    // 32-bit A 16x4: lane g=0 holds K=kb+0,1 ; g=1 holds K=kb+2,3 (B symmetric)
    f32x2 a = *(const f32x2*)(xi + kb + 2 * g);
    f32x2 b = *(const f32x2*)(xj + kb + 2 * g);
    acc = __builtin_amdgcn_wmma_f32_16x16x4_f32(false, a, false, b, (short)0, acc,
                                                false, false);
  }
  const float sqj  = sq[jBase + m];
  const int   lblj = labels[jBase + m];
  #pragma unroll
  for (int v = 0; v < 8; ++v) {
    const int i = iBase + v + 8 * g;                 // C/D: M = v + 8g, N = lane%16
    float d2 = sq[i] + sqj - 2.f * acc[v];
    float d  = sqrtf(fmaxf(d2, 1e-12f));
    bool same = (labels[i] == lblj);
    float apv = same ? d : 0.f;                      // neutral for max (d >= 0)
    float anv = same ? __builtin_inff() : d;         // neutral for min
    #pragma unroll
    for (int off = 1; off < 16; off <<= 1) {
      apv = fmaxf(apv, __shfl_xor(apv, off, 32));
      anv = fminf(anv, __shfl_xor(anv, off, 32));
    }
    if (m == 0) {  // int-bit atomics valid: all values are non-negative floats
      atomicMax(&ap[i], __float_as_int(apv));
      atomicMin(&an[i], __float_as_int(anv));
    }
  }
}

// ---------------- main GEMM: 64 rows x 128 classes per block, bf16 WMMA ----------------
__global__ __launch_bounds__(256) void k_logits(const float* __restrict__ x,
                                                const float* __restrict__ W,
                                                const float* __restrict__ bias,
                                                const int*   __restrict__ labels,
                                                float* __restrict__ pmax,
                                                float* __restrict__ psexp,
                                                float* __restrict__ psum,
                                                float* __restrict__ lablogit) {
  extern __shared__ char smem[];
  bf16_t* Xs = (bf16_t*)smem;                        // RT x XSS bf16 = 33792 B
  bf16_t* Ws = (bf16_t*)(smem + RT * XSS * 2);       // CT x WSS bf16 = 67584 B
  float*  Ls = (float*)smem;                         // aliases Xs after compute (RT*LTS*4 = 33792 B)

  const int tid      = threadIdx.x;
  const int rowBaseG = blockIdx.x * RT;              // 16 row tiles
  const int clsBaseG = blockIdx.y * CT;              // 782 class tiles
  const int w = tid >> 5, lane = tid & 31;
  const int g = lane >> 4, m = lane & 15;
  const int rSub = (w & 3) * 16;                     // wave's row sub-tile
  const int qSub = (w >> 2) * 64;                    // wave's class quad (4 x 16 cols)
  f32x8 acc[4] = {{}, {}, {}, {}};

  for (int ph = 0; ph < EMBED / KC; ++ph) {
    // ---- stage X chunk (f32 -> bf16) ----
    for (int idx = tid; idx < RT * (KC / 4); idx += 256) {
      int r = idx / (KC / 4), c4 = idx % (KC / 4);
      float4 v = *(const float4*)(x + (size_t)(rowBaseG + r) * EMBED + ph * KC + c4 * 4);
      bf16_t* dst = Xs + r * XSS + c4 * 4;
      dst[0] = (bf16_t)v.x; dst[1] = (bf16_t)v.y; dst[2] = (bf16_t)v.z; dst[3] = (bf16_t)v.w;
    }
    // ---- stage W chunk (f32 -> bf16, zero-fill OOB classes) ----
    for (int idx = tid; idx < CT * (KC / 4); idx += 256) {
      int r = idx / (KC / 4), c4 = idx % (KC / 4);
      int cg = clsBaseG + r;
      float4 v = make_float4(0.f, 0.f, 0.f, 0.f);
      if (cg < NCLS) v = *(const float4*)(W + (size_t)cg * EMBED + ph * KC + c4 * 4);
      bf16_t* dst = Ws + r * WSS + c4 * 4;
      dst[0] = (bf16_t)v.x; dst[1] = (bf16_t)v.y; dst[2] = (bf16_t)v.z; dst[3] = (bf16_t)v.w;
    }
    __syncthreads();

    // ---- prefetch next K-phase of this W tile toward L2 while WMMAs run ----
    // (global_prefetch_b8: no VGPR return, no LOADcnt -- pure latency hiding)
    if (ph + 1 < EMBED / KC) {
      for (int r = tid; r < CT; r += 256) {
        int cg = clsBaseG + r;
        if (cg < NCLS) {
          const float* nxt = W + (size_t)cg * EMBED + (ph + 1) * KC;
          #pragma unroll
          for (int q = 0; q < 8; ++q)            // 8 x 128B lines = 1024 B row chunk
            __builtin_prefetch(nxt + q * 32, 0, 1);
        }
      }
    }

    for (int kb = 0; kb < KC; kb += 32) {
      // A frag (16-bit A 16x32): lane m holds M=rSub+m; element e -> K = e + 8g + (e>=8 ? 8 : 0)
      bf16x16 a;
      const bf16_t* xrow = Xs + (rSub + m) * XSS + kb;
      #pragma unroll
      for (int e = 0; e < 16; e += 2) {
        int kl = e + 8 * g + ((e >= 8) ? 8 : 0);
        bf16x2 p = *(const bf16x2*)(xrow + kl);
        a[e] = p.x; a[e + 1] = p.y;
      }
      #pragma unroll
      for (int j = 0; j < 4; ++j) {
        // B frag (32x16): lane n=m holds column; element e -> K = kb + e + 8g + (e>=8 ? 8 : 0)
        bf16x16 bfr;
        const bf16_t* wrow = Ws + (qSub + j * 16 + m) * WSS + kb;
        #pragma unroll
        for (int e = 0; e < 16; e += 2) {
          int kl = e + 8 * g + ((e >= 8) ? 8 : 0);
          bf16x2 p = *(const bf16x2*)(wrow + kl);
          bfr[e] = p.x; bfr[e + 1] = p.y;
        }
        acc[j] = __builtin_amdgcn_wmma_f32_16x16x32_bf16(false, a, false, bfr, (short)0,
                                                         acc[j], false, false);
      }
    }
    __syncthreads();
  }

  // ---- logits + bias -> LDS tile (Ls aliases dead Xs region) ----
  #pragma unroll
  for (int j = 0; j < 4; ++j) {
    int col = qSub + j * 16 + m;
    int cg  = clsBaseG + col;
    float bb = (cg < NCLS) ? bias[cg] : 0.f;
    #pragma unroll
    for (int v = 0; v < 8; ++v) {
      int r = rSub + v + 8 * g;                      // C/D layout: M = v + 8g
      Ls[r * LTS + col] = acc[j][v] + bb;
    }
  }
  __syncthreads();

  // ---- per-row partial softmax stats over this class tile ----
  const int cn = min(CT, NCLS - clsBaseG);
  if (tid < RT) {
    const int rowG = rowBaseG + tid;
    const float* lr = Ls + tid * LTS;
    float mmax = -__builtin_inff();
    for (int j = 0; j < cn; ++j) mmax = fmaxf(mmax, lr[j]);
    float s = 0.f, t = 0.f;
    for (int j = 0; j < cn; ++j) { float v = lr[j]; s += expf(v - mmax); t += v; }
    const size_t o = (size_t)blockIdx.y * BATCH + rowG;
    pmax[o] = mmax; psexp[o] = s; psum[o] = t;
    int lo = labels[rowG] - clsBaseG;
    if (lo >= 0 && lo < cn) lablogit[rowG] = lr[lo]; // exactly one tile owns it
  }
}

// ---------------- deterministic LSE merge + per-row CE ----------------
__global__ __launch_bounds__(256) void k_ce(const float* __restrict__ pmax,
                                            const float* __restrict__ psexp,
                                            const float* __restrict__ psum,
                                            const float* __restrict__ lablogit,
                                            float* __restrict__ ce) {
  __shared__ float red[256];
  const int row = blockIdx.x, tid = threadIdx.x;
  float mmax = -__builtin_inff();
  for (int i = tid; i < NT; i += 256) mmax = fmaxf(mmax, pmax[(size_t)i * BATCH + row]);
  red[tid] = mmax; __syncthreads();
  for (int s = 128; s > 0; s >>= 1) {
    if (tid < s) red[tid] = fmaxf(red[tid], red[tid + s]);
    __syncthreads();
  }
  mmax = red[0]; __syncthreads();

  float se = 0.f, sl = 0.f;
  for (int i = tid; i < NT; i += 256) {
    const size_t o = (size_t)i * BATCH + row;
    se += psexp[o] * expf(pmax[o] - mmax);
    sl += psum[o];
  }
  red[tid] = se; __syncthreads();
  for (int s = 128; s > 0; s >>= 1) { if (tid < s) red[tid] += red[tid + s]; __syncthreads(); }
  se = red[0]; __syncthreads();
  red[tid] = sl; __syncthreads();
  for (int s = 128; s > 0; s >>= 1) { if (tid < s) red[tid] += red[tid + s]; __syncthreads(); }
  sl = red[0];

  if (tid == 0) {
    float lse = mmax + logf(se);
    ce[row] = -(1.f - EPS_S) * (lablogit[row] - lse)
              - (EPS_S / (float)NCLS) * (sl - (float)NCLS * lse);
  }
}

// ---------------- final scalar: mean hinge + mean CE ----------------
__global__ __launch_bounds__(256) void k_final(const float* __restrict__ apf,
                                               const float* __restrict__ anf,
                                               const float* __restrict__ ce,
                                               float* __restrict__ out) {
  __shared__ float red[256];
  const int tid = threadIdx.x;
  float acc = 0.f;
  for (int i = tid; i < BATCH; i += 256)
    acc += fmaxf(apf[i] - anf[i] + MARGIN, 0.f) + ce[i];
  red[tid] = acc; __syncthreads();
  for (int s = 128; s > 0; s >>= 1) { if (tid < s) red[tid] += red[tid + s]; __syncthreads(); }
  if (tid == 0) out[0] = red[0] / (float)BATCH;
}

extern "C" void kernel_launch(void* const* d_in, const int* in_sizes, int n_in,
                              void* d_out, int out_size, void* d_ws, size_t ws_size,
                              hipStream_t stream) {
  const float* x      = (const float*)d_in[0];
  const float* W      = (const float*)d_in[1];
  const float* b      = (const float*)d_in[2];
  const int*   labels = (const int*)d_in[3];

  char* ws = (char*)d_ws;
  size_t off = 0;
  auto carve = [&](size_t bytes) -> void* {
    void* p = ws + off;
    off = (off + bytes + 255) & ~(size_t)255;
    return p;
  };
  float* sq       = (float*)carve(BATCH * sizeof(float));
  int*   ap       = (int*)  carve(BATCH * sizeof(int));
  int*   an       = (int*)  carve(BATCH * sizeof(int));
  float* lablog   = (float*)carve(BATCH * sizeof(float));
  float* ce       = (float*)carve(BATCH * sizeof(float));
  float* pmax     = (float*)carve((size_t)NT * BATCH * sizeof(float));
  float* psexp    = (float*)carve((size_t)NT * BATCH * sizeof(float));
  float* psum     = (float*)carve((size_t)NT * BATCH * sizeof(float));
  (void)ws_size; (void)in_sizes; (void)n_in; (void)out_size;

  k_init<<<dim3(4), 256, 0, stream>>>(ap, an, lablog);
  k_sq  <<<dim3(BATCH), 128, 0, stream>>>(x, sq);
  k_gram<<<dim3(BATCH / 16, BATCH / 16), 32, 0, stream>>>(x, sq, labels, ap, an);

  const size_t smem = (size_t)RT * XSS * 2 + (size_t)CT * WSS * 2;  // 101376 B
  // blockIdx.x = row tile (fast) so the 16 blocks sharing a W tile run together -> L2 reuse
  k_logits<<<dim3(BATCH / RT, NT), 256, smem, stream>>>(x, W, b, labels,
                                                        pmax, psexp, psum, lablog);
  k_ce   <<<dim3(BATCH), 256, 0, stream>>>(pmax, psexp, psum, lablog, ce);
  k_final<<<dim3(1), 256, 0, stream>>>((const float*)ap, (const float*)an, ce, (float*)d_out);
}
